// LSTM_uni_80307298500883
// MI455X (gfx1250) — matrix operand order repaired and checked by reading, not compiled, not run on hardware
//
#include <hip/hip_runtime.h>
#include <hip/hip_bf16.h>

// ---------------- problem constants ----------------
#define SEQ   8192
#define EMB   300
#define OUTS  300
#define DEPTH 6
#define KPAD  320          // padded reduction dim (x part or a part), 10 x 32
#define KRECI 38           // recurrence dot chunks of 8 actually used (304 >= 300)
#define GROWS 1200         // 4 gates * OUTS
#define NT    (SEQ / 16)   // 512 N tiles
#define MT    (GROWS / 16) // 75  M tiles
#define RECTHREADS 608     // 19 wave32s; rows r and r+608 per thread

typedef __attribute__((ext_vector_type(16))) __bf16 v16bf;
typedef __attribute__((ext_vector_type(8)))  float  v8f;
typedef unsigned int v4u __attribute__((ext_vector_type(4)));
typedef int          v8i __attribute__((ext_vector_type(8)));
typedef int          v4i __attribute__((ext_vector_type(4)));

// ---------------- helpers ----------------
__device__ __forceinline__ unsigned short f2bf(float f) {
  unsigned int u = __float_as_uint(f);
  u += 0x7FFFu + ((u >> 16) & 1u);   // round-to-nearest-even
  return (unsigned short)(u >> 16);
}
// one VALU op per element: low half = shl16, high half = and-mask
__device__ __forceinline__ float bf_lo(unsigned int w) {
  return __uint_as_float(w << 16);
}
__device__ __forceinline__ float bf_hi(unsigned int w) {
  return __uint_as_float(w & 0xFFFF0000u);
}
__device__ __forceinline__ float dot8_bf16(uint4 w, const float* a8) {
  float s;
  s  = bf_lo(w.x) * a8[0];
  s += bf_hi(w.x) * a8[1];
  s += bf_lo(w.y) * a8[2];
  s += bf_hi(w.y) * a8[3];
  s += bf_lo(w.z) * a8[4];
  s += bf_hi(w.z) * a8[5];
  s += bf_lo(w.w) * a8[6];
  s += bf_hi(w.w) * a8[7];
  return s;
}

// ---------------- TDM: 1-row (1200 x f32) tile, global -> LDS ----------------
#if __has_include(<hip/amd_detail/amd_gfx1250_TDM.h>)
#define TDM_CALL(g0, g1, z4) \
  __builtin_amdgcn_tensor_load_to_lds(g0, g1, z4, z4, (v8i){0,0,0,0,0,0,0,0}, 0)
#else
#define TDM_CALL(g0, g1, z4) \
  __builtin_amdgcn_tensor_load_to_lds(g0, g1, z4, z4, 0)
#endif

__device__ __forceinline__ void tdm_load_gx_row(const float* gaddr, unsigned lds_off) {
  unsigned long long ga = (unsigned long long)(size_t)gaddr;
  // D# group 0 (ISA 08 §8.3): count=1 | lds_addr | global_addr[56:0] | type=2
  v4u g0;
  g0.x = 1u;                                                // count = 1 valid descriptor
  g0.y = lds_off;                                           // lds_addr (bytes)
  g0.z = (unsigned)(ga & 0xFFFFFFFFu);                      // global_addr[31:0]
  g0.w = (unsigned)((ga >> 32) & 0x01FFFFFFu) | (2u << 30); // global_addr[56:32] | type=2
  // D# group 1 (ISA 08 §8.4)
  v8i g1;
  g1[0] = (int)(2u << 16);                 // workgroup_mask=0, data_size=2 (4 bytes)
  g1[1] = (int)((GROWS & 0xFFFFu) << 16);  // atomic_barrier_addr=0 | tensor_dim0[15:0]
  g1[2] = (int)((GROWS >> 16) | (1u << 16)); // tensor_dim0[31:16] | tensor_dim1[15:0]=1
  g1[3] = (int)((unsigned)GROWS << 16);    // tensor_dim1[31:16]=0 | tile_dim0=1200
  g1[4] = 1;                               // tile_dim1=1 | tile_dim2=0
  g1[5] = GROWS;                           // tensor_dim0_stride[31:0]
  g1[6] = 0;                               // stride0[47:32] | stride1[15:0]
  g1[7] = 0;                               // stride1[47:16]
  v4i z4 = {0, 0, 0, 0};                   // groups 2/3 unused (<=2D tensor)
  TDM_CALL(g0, g1, z4);
}

// ---------------- kernel 1: embedding gather -> bf16 padded [SEQ][KPAD] ----------------
__global__ __launch_bounds__(KPAD) void embed_kernel(
    const int* __restrict__ x_i, const float* __restrict__ emb,
    unsigned short* __restrict__ xbf) {
  const int t = blockIdx.x;
  const int k = threadIdx.x;
  const int idx = x_i[t];
  unsigned short v = 0;
  if (k < EMB) v = f2bf(emb[(size_t)idx * EMB + k]);
  xbf[(size_t)t * KPAD + k] = v;   // pad columns stay zero forever
}

// ---------------- kernel 2: per-layer weight prep (split Wx | Wa, fp32->bf16) ----------------
__global__ __launch_bounds__(KPAD) void wprep_kernel(
    const float* __restrict__ wc, const float* __restrict__ wu,
    const float* __restrict__ wf, const float* __restrict__ wo,
    int l, unsigned short* __restrict__ Wx, unsigned short* __restrict__ Wa) {
  const int r = blockIdx.x;          // gate row 0..1199
  const int k = threadIdx.x;         // 0..319
  const int g = r / OUTS;            // 0=cand,1=u,2=f,3=o
  const int i = r % OUTS;
  const float* W = (g == 0) ? wc : (g == 1) ? wu : (g == 2) ? wf : wo;
  const float* row = W + ((size_t)l * OUTS + i) * (2 * EMB);   // IN = 600
  unsigned short vx = 0, va = 0;
  if (k < EMB) { vx = f2bf(row[k]); va = f2bf(row[EMB + k]); }
  Wx[(size_t)r * KPAD + k] = vx;
  Wa[(size_t)r * KPAD + k] = va;
}

// ---------------- kernel 3: WMMA GEMM  Gx[t][r] = sum_k Wx[r][k] * X[t][k] ----------------
__global__ __launch_bounds__(256) void lstm_gemm_wmma(
    const unsigned short* __restrict__ A,   // Wx [GROWS][KPAD] bf16
    const unsigned short* __restrict__ B,   // xbf [SEQ][KPAD] bf16
    float* __restrict__ Gx) {               // [SEQ][GROWS] f32 (transposed store)
  const int wid  = threadIdx.x >> 5;
  const int lane = threadIdx.x & 31;
  const int tile = blockIdx.x * 8 + wid;   // exact fill: 4800 blocks * 8 waves = 75*512
  if (tile >= MT * NT) return;             // wave-uniform guard (EXEC all-ones inside)
  const int mt = tile % MT;
  const int nt = tile / MT;
  const int lo = lane & 15;
  const int hi = lane >> 4;

  // A lane layout (ISA 7.12.2): elem j<8 -> K = kk + hi*8 + j ; j>=8 -> K = kk + 16 + hi*8 + (j-8)
  const unsigned short* arow = A + (size_t)(mt * 16 + lo) * KPAD + hi * 8;
  // B lane layout: same as A: elem j<8 -> K = kk + hi*8 + j ; j>=8 -> K = kk + 16 + hi*8 + (j-8)
  const unsigned short* brow = B + (size_t)(nt * 16 + lo) * KPAD + hi * 8;

  v8f acc = {};
#pragma unroll
  for (int kk = 0; kk < KPAD; kk += 32) {
    union { v16bf v; uint4 q[2]; } a, b;
    a.q[0] = *(const uint4*)(arow + kk);
    a.q[1] = *(const uint4*)(arow + kk + 16);
    b.q[0] = *(const uint4*)(brow + kk);
    b.q[1] = *(const uint4*)(brow + kk + 16);
    acc = __builtin_amdgcn_wmma_f32_16x16x32_bf16(
        /*neg_a=*/false, a.v, /*neg_b=*/false, b.v,
        /*c_mod=*/(short)0, acc, /*reuse_a=*/false, /*reuse_b=*/false);
  }
  // D: VGPR r -> M = mt*16 + r + 8*hi, N = nt*16 + lo. Store transposed: Gx[n][m].
  float* dp = Gx + (size_t)(nt * 16 + lo) * GROWS + mt * 16 + hi * 8;
  *(float4*)(dp)     = make_float4(acc[0], acc[1], acc[2], acc[3]);
  *(float4*)(dp + 4) = make_float4(acc[4], acc[5], acc[6], acc[7]);
}

// ---------------- kernel 4: sequential recurrence (one workgroup, TDM-fed) ----------------
__global__ __launch_bounds__(RECTHREADS) void lstm_rec(
    const float* __restrict__ Gx,              // [SEQ][GROWS] precomputed Wx @ x
    const unsigned short* __restrict__ Wa,     // [GROWS][KPAD] bf16
    const float* __restrict__ bc, const float* __restrict__ bu,
    const float* __restrict__ bfr, const float* __restrict__ bo,   // + l*OUTS
    const float* __restrict__ a0, const float* __restrict__ c0,    // + l*OUTS
    unsigned short* __restrict__ xbf,          // next layer's B operand (non-last)
    float* __restrict__ xout,                  // d_out x region (last layer)
    float* __restrict__ afin, float* __restrict__ cfin,
    int last) {
  __shared__ float a_sh[2][KPAD];
  __shared__ float c_sh[OUTS];
  __shared__ float g_sh[GROWS];
  __shared__ float gx_sh[2][GROWS];            // TDM ping-pong landing zone

  const int tid = threadIdx.x;
  const int r0 = tid;
  const int r1 = tid + RECTHREADS;
  const bool has1 = (r1 < GROWS);
  const bool tdm_wave = (tid < 32);            // one wave issues/waits TDM

  for (int i = tid; i < KPAD; i += RECTHREADS) {
    a_sh[0][i] = (i < OUTS) ? a0[i] : 0.0f;
    a_sh[1][i] = 0.0f;                         // pad stays zero in both buffers
  }
  if (tid < OUTS) c_sh[tid] = c0[tid];

  // hoist biases
  const int g0i = r0 / OUTS, g0j = r0 % OUTS;
  const float* bt0 = (g0i == 0) ? bc : (g0i == 1) ? bu : (g0i == 2) ? bfr : bo;
  const float bias0 = bt0[g0j];
  float bias1 = 0.0f;
  if (has1) {
    const int g1i = r1 / OUTS, g1j = r1 % OUTS;
    const float* bt1 = (g1i == 0) ? bc : (g1i == 1) ? bu : (g1i == 2) ? bfr : bo;
    bias1 = bt1[g1j];
  }

  const uint4* w0 = (const uint4*)(Wa + (size_t)r0 * KPAD);
  const uint4* w1 = (const uint4*)(Wa + (size_t)(has1 ? r1 : r0) * KPAD);

  const unsigned gx_lds0 = (unsigned)(size_t)(&gx_sh[0][0]);
  const unsigned gx_lds1 = (unsigned)(size_t)(&gx_sh[1][0]);

  // prologue: DMA gate row 0 into buffer 0
  if (tdm_wave) tdm_load_gx_row(Gx, gx_lds0);

  __syncthreads();

  for (int t = 0; t < SEQ; ++t) {
    if (tdm_wave) __builtin_amdgcn_s_wait_tensorcnt(0);   // row t landed in LDS
    __syncthreads();                                      // ... and visible to all waves
    if (tdm_wave && (t + 1 < SEQ))                        // overlap next row with matvec
      tdm_load_gx_row(Gx + (size_t)(t + 1) * GROWS, ((t + 1) & 1) ? gx_lds1 : gx_lds0);

    const float* ap = a_sh[t & 1];
    const float* gr = gx_sh[t & 1];

    float acc0 = 0.0f, acc1 = 0.0f;
#pragma unroll 2
    for (int it = 0; it < KRECI; ++it) {       // 38 chunks of 8 bf16 MACs (K=304)
      const uint4 wA = w0[it];
      const uint4 wB = w1[it];
      const float* a8 = ap + it * 8;
      acc0 += dot8_bf16(wA, a8);
      acc1 += dot8_bf16(wB, a8);
    }
    g_sh[r0] = gr[r0] + bias0 + acc0;
    if (has1) g_sh[r1] = gr[r1] + bias1 + acc1;
    __syncthreads();

    if (tid < OUTS) {
      const float cand = tanhf(g_sh[tid]);
      const float su = 1.0f / (1.0f + expf(-g_sh[OUTS + tid]));
      const float sf = 1.0f / (1.0f + expf(-g_sh[2 * OUTS + tid]));
      const float so = 1.0f / (1.0f + expf(-g_sh[3 * OUTS + tid]));
      const float cn = su * cand + sf * c_sh[tid];
      const float an = so * tanhf(cn);
      c_sh[tid] = cn;
      a_sh[(t + 1) & 1][tid] = an;
      if (last) xout[(size_t)t * OUTS + tid] = an;          // final f32 output
      else      xbf[(size_t)t * KPAD + tid] = f2bf(an);     // next layer GEMM operand
    }
    __syncthreads();
  }

  if (tid < OUTS) {
    afin[tid] = a_sh[SEQ & 1][tid];
    cfin[tid] = c_sh[tid];
  }
}

// ---------------- host launcher ----------------
extern "C" void kernel_launch(void* const* d_in, const int* in_sizes, int n_in,
                              void* d_out, int out_size, void* d_ws, size_t ws_size,
                              hipStream_t stream) {
  (void)in_sizes; (void)n_in; (void)out_size; (void)ws_size;
  const int*   x_i = (const int*)  d_in[0];
  const float* a_0 = (const float*)d_in[1];
  const float* c_0 = (const float*)d_in[2];
  const float* emb = (const float*)d_in[3];
  const float* w_c = (const float*)d_in[4];
  const float* w_u = (const float*)d_in[5];
  const float* w_f = (const float*)d_in[6];
  const float* w_o = (const float*)d_in[7];
  const float* b_c = (const float*)d_in[8];
  const float* b_u = (const float*)d_in[9];
  const float* b_f = (const float*)d_in[10];
  const float* b_o = (const float*)d_in[11];
  float* out = (float*)d_out;

  // workspace layout (256B-aligned offsets)
  char* ws = (char*)d_ws;
  const size_t off_xbf = 0;                                   // SEQ*KPAD bf16 = 5,242,880
  const size_t off_wx  = off_xbf + (size_t)SEQ * KPAD * 2;    // GROWS*KPAD bf16 = 768,000
  const size_t off_wa  = off_wx  + (size_t)GROWS * KPAD * 2;
  const size_t off_gx  = off_wa  + (size_t)GROWS * KPAD * 2;  // SEQ*GROWS f32 = 39,321,600
  unsigned short* xbf = (unsigned short*)(ws + off_xbf);
  unsigned short* Wx  = (unsigned short*)(ws + off_wx);
  unsigned short* Wa  = (unsigned short*)(ws + off_wa);
  float*          Gx  = (float*)        (ws + off_gx);

  float* xout = out;                           // [SEQ][OUTS]
  float* afin = out + (size_t)SEQ * OUTS;      // [DEPTH][OUTS]
  float* cfin = afin + DEPTH * OUTS;           // [DEPTH][OUTS]

  embed_kernel<<<SEQ, KPAD, 0, stream>>>(x_i, emb, xbf);

  for (int l = 0; l < DEPTH; ++l) {
    wprep_kernel<<<GROWS, KPAD, 0, stream>>>(w_c, w_u, w_f, w_o, l, Wx, Wa);
    lstm_gemm_wmma<<<(MT * NT) / 8, 256, 0, stream>>>(Wx, xbf, Gx);
    const int last = (l == DEPTH - 1);
    lstm_rec<<<1, RECTHREADS, 0, stream>>>(
        Gx, Wa,
        b_c + (size_t)l * OUTS, b_u + (size_t)l * OUTS,
        b_f + (size_t)l * OUTS, b_o + (size_t)l * OUTS,
        a_0 + (size_t)l * OUTS, c_0 + (size_t)l * OUTS,
        xbf, xout, afin + (size_t)l * OUTS, cfin + (size_t)l * OUTS, last);
  }
}
